// MambaLayer_26104811225259
// MI455X (gfx1250) — matrix operand order repaired and checked by reading, not compiled, not run on hardware
//
#include <hip/hip_runtime.h>
#include <hip/hip_bf16.h>

typedef __attribute__((ext_vector_type(16))) _Float16 v16h;
typedef __attribute__((ext_vector_type(8)))  _Float16 v8h;
typedef __attribute__((ext_vector_type(8)))  float    v8f;
typedef __attribute__((ext_vector_type(4)))  unsigned int u32x4;
typedef __attribute__((ext_vector_type(8)))  int          i32x8;
typedef __attribute__((ext_vector_type(4)))  int          i32x4;

#define DIMC   512
#define LSEQ   4096
#define BSZ    4
#define DINNER 1024
#define DSTATE 16
#define DTRANK 32
#define NROWS  (BSZ * LSEQ)     // 16384 token rows

#define LDT    40               // padded LDS row stride in halves (64B data + 16B pad)

#if defined(__has_builtin)
#if __has_builtin(__builtin_amdgcn_tensor_load_to_lds)
#define HAVE_TDM 1
#endif
#endif
#ifndef HAVE_TDM
#define HAVE_TDM 0
#endif

// ---------------------------------------------------------------------------
// TDM: DMA one 2D f16 tile (tile_w=32 elems x tile_rows) global -> LDS with
// hardware row padding (16 DWORDs data, then 4 DWORDs pad -> 80B LDS rows).
// Issued once per workgroup (wave 0); tracked by TENSORcnt.
// ---------------------------------------------------------------------------
#if HAVE_TDM
__device__ __forceinline__ void tdm_tile_load(unsigned int lds_off_bytes,
                                              const _Float16* gsrc,
                                              unsigned int row_stride_elems,
                                              unsigned int tile_rows) {
    unsigned long long ga = (unsigned long long)(size_t)gsrc;
    u32x4 g0;
    g0[0] = 1u;                                              // count=1, user desc
    g0[1] = lds_off_bytes;                                   // lds_addr
    g0[2] = (unsigned int)ga;                                // global_addr[31:0]
    g0[3] = (unsigned int)((ga >> 32) & 0x01FFFFFFu) | (2u << 30); // addr[56:32] | type=2
    const unsigned int td0 = 0x7FFFFFFFu, td1 = 0x7FFFFFFFu; // huge tensor dims (no OOB)
    i32x8 g1;
    g1[0] = (int)((1u << 16)        // data_size = 2 bytes
                | (1u << 20)        // pad_enable
                | (3u << 22)        // pad_interval: every 16 DWORDs (64B row)
                | (3u << 25));      // pad_amount: 4 DWORDs (16B)
    g1[1] = (int)((td0 & 0xFFFFu) << 16);                          // tensor_dim0[15:0]
    g1[2] = (int)(((td0 >> 16) & 0xFFFFu) | ((td1 & 0xFFFFu) << 16));
    g1[3] = (int)(((td1 >> 16) & 0xFFFFu) | (32u << 16));          // tile_dim0 = 32 elems
    g1[4] = (int)(tile_rows & 0xFFFFu);                            // tile_dim1; tile_dim2=0
    g1[5] = (int)row_stride_elems;                                 // dim0_stride[31:0]
    g1[6] = 0;                                                     // dim0_stride[47:32], dim1_stride lo
    g1[7] = 0;
    i32x4 z4 = {};
#if __clang_major__ >= 23
    i32x8 z8 = {};
    __builtin_amdgcn_tensor_load_to_lds(g0, g1, z4, z4, z8, 0);
#else
    __builtin_amdgcn_tensor_load_to_lds(g0, g1, z4, z4, 0);
#endif
}
#endif

// ---------------------------------------------------------------------------
// fp32 -> fp16 weight conversion
// ---------------------------------------------------------------------------
__global__ void cvt_f32_f16(const float* __restrict__ src, _Float16* __restrict__ dst, int n) {
    int i = blockIdx.x * 256 + threadIdx.x;
    if (i < n) dst[i] = (_Float16)src[i];
}

// ---------------------------------------------------------------------------
// LayerNorm over DIM for each (b,l) row of x^T; writes f16 xn.
// ---------------------------------------------------------------------------
__global__ __launch_bounds__(512) void ln_kernel(const float* __restrict__ x,
                                                 const float* __restrict__ w,
                                                 const float* __restrict__ bv,
                                                 _Float16* __restrict__ xn) {
    __shared__ float tile[16 * 513];
    const int b  = blockIdx.x >> 8;
    const int l0 = (blockIdx.x & 255) << 4;
    const float* xb = x + (size_t)b * DIMC * LSEQ;
    #pragma unroll
    for (int i = 0; i < 16; ++i) {
        int idx = threadIdx.x + i * 512;
        int li = idx & 15, k = idx >> 4;
        tile[li * 513 + k] = xb[(size_t)k * LSEQ + l0 + li];
    }
    __syncthreads();
    const int wave = threadIdx.x >> 5, lane = threadIdx.x & 31;
    float s = 0.f, s2 = 0.f;
    #pragma unroll
    for (int i = 0; i < 16; ++i) {
        float v = tile[wave * 513 + i * 32 + lane];
        s += v; s2 += v * v;
    }
    #pragma unroll
    for (int off = 16; off >= 1; off >>= 1) {
        s  += __shfl_xor(s,  off, 32);
        s2 += __shfl_xor(s2, off, 32);
    }
    const float mu   = s * (1.f / 512.f);
    const float rstd = rsqrtf(s2 * (1.f / 512.f) - mu * mu + 1e-5f);
    const size_t row = (size_t)(b * LSEQ + l0 + wave) * DIMC;
    #pragma unroll
    for (int i = 0; i < 16; ++i) {
        int k = i * 32 + lane;
        float v = (tile[wave * 513 + k] - mu) * rstd * w[k] + bv[k];
        xn[row + k] = (_Float16)v;
    }
}

// ---------------------------------------------------------------------------
// Fragment helpers (layouts per CDNA5 ISA 7.12.2)
// ---------------------------------------------------------------------------
__device__ __forceinline__ v16h frag_a(const _Float16* base, int row, int lh) {
    v8h lo = *(const v8h*)&base[row * LDT + lh * 8];
    v8h hi = *(const v8h*)&base[row * LDT + 16 + lh * 8];
    return __builtin_shufflevector(lo, hi, 0,1,2,3,4,5,6,7,8,9,10,11,12,13,14,15);
}
__device__ __forceinline__ v16h frag_b(const _Float16* base, int col, int lh) {
    v8h lo = *(const v8h*)&base[col * LDT + lh * 8];
    v8h hi = *(const v8h*)&base[col * LDT + 16 + lh * 8];
    return __builtin_shufflevector(lo, hi, 0,1,2,3,4,5,6,7,8,9,10,11,12,13,14,15);
}

// ---------------------------------------------------------------------------
// WMMA GEMM: out[m,n] = sum_k A[m,k] * W[n,k]
// Block tile 128x128, 8 waves (2 x 4), wave tile 64x32 -> 4x2 WMMA accums.
// TDM double-buffered LDS staging (DMA of next K-slab overlaps compute).
// MODE 0: split-store f16 halves (xin | z).  MODE 2: fp32 transposed store.
// ---------------------------------------------------------------------------
template <int MODE>
__global__ __launch_bounds__(256) void gemm_bt128(const _Float16* __restrict__ A,
                                                  const _Float16* __restrict__ W,
                                                  int K,
                                                  _Float16* __restrict__ oh0,
                                                  _Float16* __restrict__ oh1,
                                                  float* __restrict__ of) {
    __shared__ __align__(16) _Float16 sA[2][128 * LDT];
    __shared__ __align__(16) _Float16 sB[2][128 * LDT];
    const int tid  = threadIdx.x;
    const int m0   = blockIdx.y * 128;
    const int n0   = blockIdx.x * 128;
    const int wave = tid >> 5, lane = tid & 31;
    const int wm = wave >> 2, wn = wave & 3;
    const int l16 = lane & 15, lh = lane >> 4;

    v8f acc[4][2];
    #pragma unroll
    for (int i = 0; i < 4; ++i)
        #pragma unroll
        for (int j = 0; j < 2; ++j) acc[i][j] = v8f{};

#if HAVE_TDM
    const bool issuer = (wave == 0);
    const unsigned int ldsA = (unsigned int)(size_t)&sA[0][0];
    const unsigned int ldsB = (unsigned int)(size_t)&sB[0][0];
    const unsigned int bufBytes = 128 * LDT * 2;
    if (issuer) {
        tdm_tile_load(ldsA, &A[(size_t)m0 * K], K, 128);
        tdm_tile_load(ldsB, &W[(size_t)n0 * K], K, 128);
    }
    for (int k0 = 0; k0 < K; k0 += 32) {
        const int cur = (k0 >> 5) & 1;
        const bool has_next = (k0 + 32) < K;
        if (issuer) {
            if (has_next) {
                tdm_tile_load(ldsA + (cur ^ 1) * bufBytes, &A[(size_t)m0 * K + k0 + 32], K, 128);
                tdm_tile_load(ldsB + (cur ^ 1) * bufBytes, &W[(size_t)n0 * K + k0 + 32], K, 128);
                __builtin_amdgcn_s_wait_tensorcnt(2);   // current slab complete
            } else {
                __builtin_amdgcn_s_wait_tensorcnt(0);
            }
        }
        __syncthreads();                                 // publish DMA'd tiles
        const _Float16* bA = sA[cur];
        const _Float16* bB = sB[cur];
#else
    for (int k0 = 0; k0 < K; k0 += 32) {
        __syncthreads();
        #pragma unroll
        for (int c = tid; c < 512; c += 256) {
            int row = c >> 2, cc = c & 3;
            *(v8h*)&sA[0][row * LDT + cc * 8] = *(const v8h*)&A[(size_t)(m0 + row) * K + k0 + cc * 8];
            *(v8h*)&sB[0][row * LDT + cc * 8] = *(const v8h*)&W[(size_t)(n0 + row) * K + k0 + cc * 8];
        }
        __syncthreads();
        const _Float16* bA = sA[0];
        const _Float16* bB = sB[0];
#endif
        v16h af[4], bf[2];
        #pragma unroll
        for (int i = 0; i < 4; ++i) af[i] = frag_a(bA, wm * 64 + i * 16 + l16, lh);
        #pragma unroll
        for (int j = 0; j < 2; ++j) bf[j] = frag_b(bB, wn * 32 + j * 16 + l16, lh);
        #pragma unroll
        for (int i = 0; i < 4; ++i)
            #pragma unroll
            for (int j = 0; j < 2; ++j)
                acc[i][j] = __builtin_amdgcn_wmma_f32_16x16x32_f16(
                    false, af[i], false, bf[j], (short)0, acc[i][j], false, false);
#if HAVE_TDM
        __syncthreads();                                 // done reading buf before re-DMA
#endif
    }
    // Epilogue. C/D layout: lane%16 = N, vgpr r = M (+8 for lanes 16-31).
    #pragma unroll
    for (int i = 0; i < 4; ++i) {
        int mb = m0 + wm * 64 + i * 16 + lh * 8;
        #pragma unroll
        for (int j = 0; j < 2; ++j) {
            int col = n0 + wn * 32 + j * 16 + l16;
            #pragma unroll
            for (int r = 0; r < 8; ++r) {
                int m = mb + r;
                float v = acc[i][j][r];
                if (MODE == 0) {
                    if (col < DINNER) oh0[(size_t)m * DINNER + col] = (_Float16)v;
                    else              oh1[(size_t)m * DINNER + col - DINNER] = (_Float16)v;
                } else {
                    int bb = m >> 12, l = m & (LSEQ - 1);
                    of[((size_t)bb * DIMC + col) * LSEQ + l] = v;
                }
            }
        }
    }
}

// ---------------------------------------------------------------------------
// WMMA GEMM with N=64 (x-projection): dbl = u @ W_xproj^T, fp32 out.
// ---------------------------------------------------------------------------
__global__ __launch_bounds__(256) void gemm_n64(const _Float16* __restrict__ A,
                                                const _Float16* __restrict__ W,
                                                float* __restrict__ out) {
    __shared__ __align__(16) _Float16 sA[2][128 * LDT];
    __shared__ __align__(16) _Float16 sB[2][64 * LDT];
    const int tid = threadIdx.x;
    const int m0  = blockIdx.x * 128;
    const int wave = tid >> 5, lane = tid & 31;
    const int l16 = lane & 15, lh = lane >> 4;
    const int K = DINNER;

    v8f acc[4];
    #pragma unroll
    for (int j = 0; j < 4; ++j) acc[j] = v8f{};

#if HAVE_TDM
    const bool issuer = (wave == 0);
    const unsigned int ldsA = (unsigned int)(size_t)&sA[0][0];
    const unsigned int ldsB = (unsigned int)(size_t)&sB[0][0];
    if (issuer) {
        tdm_tile_load(ldsA, &A[(size_t)m0 * K], K, 128);
        tdm_tile_load(ldsB, W, K, 64);
    }
    for (int k0 = 0; k0 < K; k0 += 32) {
        const int cur = (k0 >> 5) & 1;
        const bool has_next = (k0 + 32) < K;
        if (issuer) {
            if (has_next) {
                tdm_tile_load(ldsA + (cur ^ 1) * (128 * LDT * 2), &A[(size_t)m0 * K + k0 + 32], K, 128);
                tdm_tile_load(ldsB + (cur ^ 1) * (64 * LDT * 2), &W[k0 + 32], K, 64);
                __builtin_amdgcn_s_wait_tensorcnt(2);
            } else {
                __builtin_amdgcn_s_wait_tensorcnt(0);
            }
        }
        __syncthreads();
        const _Float16* bA = sA[cur];
        const _Float16* bB = sB[cur];
#else
    for (int k0 = 0; k0 < K; k0 += 32) {
        __syncthreads();
        #pragma unroll
        for (int c = tid; c < 512; c += 256) {
            int row = c >> 2, cc = c & 3;
            *(v8h*)&sA[0][row * LDT + cc * 8] = *(const v8h*)&A[(size_t)(m0 + row) * K + k0 + cc * 8];
        }
        if (tid < 256) {
            int row = tid >> 2, cc = tid & 3;
            *(v8h*)&sB[0][row * LDT + cc * 8] = *(const v8h*)&W[(size_t)row * K + k0 + cc * 8];
        }
        __syncthreads();
        const _Float16* bA = sA[0];
        const _Float16* bB = sB[0];
#endif
        v16h a = frag_a(bA, wave * 16 + l16, lh);
        #pragma unroll
        for (int j = 0; j < 4; ++j) {
            v16h bfr = frag_b(bB, j * 16 + l16, lh);
            acc[j] = __builtin_amdgcn_wmma_f32_16x16x32_f16(
                false, a, false, bfr, (short)0, acc[j], false, false);
        }
#if HAVE_TDM
        __syncthreads();
#endif
    }
    int mb = m0 + wave * 16 + lh * 8;
    #pragma unroll
    for (int j = 0; j < 4; ++j) {
        int col = j * 16 + l16;
        #pragma unroll
        for (int r = 0; r < 8; ++r)
            out[(size_t)(mb + r) * 64 + col] = acc[j][r];
    }
}

// ---------------------------------------------------------------------------
// Depthwise causal conv (K=4) + SiLU, f16 in/out, fp32 math.
// ---------------------------------------------------------------------------
__global__ void conv_silu_kernel(const _Float16* __restrict__ xin,
                                 const float* __restrict__ cw,
                                 const float* __restrict__ cb,
                                 _Float16* __restrict__ u) {
    int idx = blockIdx.x * 256 + threadIdx.x;
    int d = idx & (DINNER - 1);
    int m = idx >> 10;
    int b = m >> 12, l = m & (LSEQ - 1);
    float s = cb[d];
    #pragma unroll
    for (int j = 0; j < 4; ++j) {
        int ll = l - 3 + j;
        if (ll >= 0)
            s += (float)xin[(size_t)(b * LSEQ + ll) * DINNER + d] * cw[d * 4 + j];
    }
    float us = s / (1.f + __expf(-s));
    u[(size_t)m * DINNER + d] = (_Float16)us;
}

// ---------------------------------------------------------------------------
// delta = softplus(dt @ W_dt^T + b_dt)
// ---------------------------------------------------------------------------
__global__ __launch_bounds__(256) void delta_kernel(const float* __restrict__ dbl,
                                                    const float* __restrict__ Wdt,
                                                    const float* __restrict__ bdt,
                                                    float* __restrict__ delta) {
    int m = blockIdx.x >> 2;
    int d = ((blockIdx.x & 3) << 8) + threadIdx.x;
    __shared__ float dt[DTRANK];
    if (threadIdx.x < DTRANK) dt[threadIdx.x] = dbl[(size_t)m * 64 + threadIdx.x];
    __syncthreads();
    float s = bdt[d];
    #pragma unroll
    for (int r = 0; r < DTRANK; ++r) s += dt[r] * Wdt[d * DTRANK + r];
    delta[(size_t)m * DINNER + d] = (s > 20.f) ? s : __logf(1.f + __expf(s));
}

// ---------------------------------------------------------------------------
// Selective scan: one lane per (b, d); 16 states in VGPRs; serial over L.
// ---------------------------------------------------------------------------
__global__ __launch_bounds__(256) void scan_kernel(const float* __restrict__ delta,
                                                   const _Float16* __restrict__ u,
                                                   const float* __restrict__ dbl,
                                                   const _Float16* __restrict__ z,
                                                   const float* __restrict__ A_log,
                                                   const float* __restrict__ Dv,
                                                   _Float16* __restrict__ yg) {
    const int b = blockIdx.x >> 2;
    const int d = ((blockIdx.x & 3) << 8) + threadIdx.x;
    float Ar[DSTATE], h[DSTATE];
    #pragma unroll
    for (int s = 0; s < DSTATE; ++s) {
        Ar[s] = -__expf(A_log[d * DSTATE + s]);
        h[s]  = 0.f;
    }
    const float Dd = Dv[d];
    const size_t base = (size_t)b * LSEQ;
    for (int l = 0; l < LSEQ; ++l) {
        const size_t m = base + l;
        const float  dt = delta[m * DINNER + d];
        const float  uu = (float)u[m * DINNER + d];
        const float  du = dt * uu;
        const float* bc = dbl + m * 64 + DTRANK;
        float y = 0.f;
        #pragma unroll
        for (int s = 0; s < DSTATE; ++s) {
            float hn = __expf(dt * Ar[s]) * h[s] + du * bc[s];
            h[s] = hn;
            y += hn * bc[DSTATE + s];
        }
        float zv = (float)z[m * DINNER + d];
        float sz = zv / (1.f + __expf(-zv));
        yg[m * DINNER + d] = (_Float16)((y + uu * Dd) * sz);
    }
}

// ---------------------------------------------------------------------------
extern "C" void kernel_launch(void* const* d_in, const int* in_sizes, int n_in,
                              void* d_out, int out_size, void* d_ws, size_t ws_size,
                              hipStream_t stream) {
    const float* x      = (const float*)d_in[0];
    const float* ln_w   = (const float*)d_in[1];
    const float* ln_b   = (const float*)d_in[2];
    const float* W_in   = (const float*)d_in[3];
    const float* conv_w = (const float*)d_in[4];
    const float* conv_b = (const float*)d_in[5];
    const float* W_xprj = (const float*)d_in[6];
    const float* W_dt   = (const float*)d_in[7];
    const float* b_dt   = (const float*)d_in[8];
    const float* A_log  = (const float*)d_in[9];
    const float* Dvec   = (const float*)d_in[10];
    const float* W_out  = (const float*)d_in[11];
    float* out = (float*)d_out;

    char* ws = (char*)d_ws;
    auto alloc = [&](size_t bytes) -> char* {
        char* p = ws;
        ws += (bytes + 255) & ~(size_t)255;
        return p;
    };
    _Float16* xn_h   = (_Float16*)alloc((size_t)NROWS * DIMC * 2);
    _Float16* Win_h  = (_Float16*)alloc((size_t)2 * DINNER * DIMC * 2);
    _Float16* Wxp_h  = (_Float16*)alloc((size_t)64 * DINNER * 2);
    _Float16* Wout_h = (_Float16*)alloc((size_t)DIMC * DINNER * 2);
    _Float16* xin_h  = (_Float16*)alloc((size_t)NROWS * DINNER * 2);
    _Float16* z_h    = (_Float16*)alloc((size_t)NROWS * DINNER * 2);
    _Float16* u_h    = (_Float16*)alloc((size_t)NROWS * DINNER * 2);
    float*    dbl    = (float*)   alloc((size_t)NROWS * 64 * 4);
    float*    delta  = (float*)   alloc((size_t)NROWS * DINNER * 4);
    _Float16* yg_h   = (_Float16*)alloc((size_t)NROWS * DINNER * 2);

    cvt_f32_f16<<<(2 * DINNER * DIMC + 255) / 256, 256, 0, stream>>>(W_in, Win_h, 2 * DINNER * DIMC);
    cvt_f32_f16<<<(64 * DINNER + 255) / 256, 256, 0, stream>>>(W_xprj, Wxp_h, 64 * DINNER);
    cvt_f32_f16<<<(DIMC * DINNER + 255) / 256, 256, 0, stream>>>(W_out, Wout_h, DIMC * DINNER);

    ln_kernel<<<BSZ * (LSEQ / 16), 512, 0, stream>>>(x, ln_w, ln_b, xn_h);

    gemm_bt128<0><<<dim3(2 * DINNER / 128, NROWS / 128), 256, 0, stream>>>(
        xn_h, Win_h, DIMC, xin_h, z_h, nullptr);

    conv_silu_kernel<<<(NROWS * DINNER) / 256, 256, 0, stream>>>(xin_h, conv_w, conv_b, u_h);

    gemm_n64<<<NROWS / 128, 256, 0, stream>>>(u_h, Wxp_h, dbl);

    delta_kernel<<<NROWS * 4, 256, 0, stream>>>(dbl, W_dt, b_dt, delta);

    scan_kernel<<<(BSZ * DINNER) / 256, 256, 0, stream>>>(delta, u_h, dbl, z_h, A_log, Dvec, yg_h);

    gemm_bt128<2><<<dim3(DIMC / 128, NROWS / 128), 256, 0, stream>>>(
        yg_h, Wout_h, DINNER, nullptr, nullptr, out);
}
